// TwoWayAttentionBlock_3092376453248
// MI455X (gfx1250) — matrix operand order repaired and checked by reading, not compiled, not run on hardware
//
#include <hip/hip_runtime.h>
#include <hip/hip_bf16.h>
#include <math.h>

// ---------------------------------------------------------------------------
// TwoWayAttentionBlock for MI455X (gfx1250, wave32, WMMA f32_16x16x32_f16)
// B=4, N=1024, C=768, H=6, D=128, MLP=2304
// Workspace requirement: 18 * 4096*768 floats = ~227 MB.
// ---------------------------------------------------------------------------

typedef __attribute__((ext_vector_type(16))) _Float16 v16h;
typedef __attribute__((ext_vector_type(8)))  _Float16 v8h;
typedef __attribute__((ext_vector_type(8)))  float    v8f;

#define CDIM 768
#define HEADS 6
#define DHEAD 128

__device__ __forceinline__ float gelu_exact(float x) {
    return 0.5f * x * (1.0f + erff(x * 0.70710678118654752f));
}

// ===========================================================================
// Elementwise prep:  out[(b*ldo + off + n)*C + c] = x + (y?) + (f[c]?)
// x,y are token-major [rows][C]; rows = #tokens processed; nb = tokens/batch.
// ===========================================================================
__global__ void prep_kernel(float* __restrict__ out,
                            const float* __restrict__ x,
                            const float* __restrict__ y,
                            const float* __restrict__ f,
                            int nb, int ldo, int off) {
    unsigned idx = blockIdx.x * 256u + threadIdx.x;
    unsigned c = idx % CDIM;
    unsigned r = idx / CDIM;
    unsigned b = r / (unsigned)nb;
    unsigned n = r - b * (unsigned)nb;
    float v = x[idx];
    if (y) v += y[idx];
    if (f) v += f[c];
    out[((size_t)b * ldo + off + n) * CDIM + c] = v;
}

// ===========================================================================
// LayerNorm over C=768: out = LN(a + res) * w + b   (res nullable)
// one block (256 threads) per token
// ===========================================================================
__global__ __launch_bounds__(256) void ln_kernel(float* __restrict__ out,
                                                 const float* __restrict__ a,
                                                 const float* __restrict__ res,
                                                 const float* __restrict__ w,
                                                 const float* __restrict__ bv) {
    __shared__ float rs[256], rs2[256];
    size_t base = (size_t)blockIdx.x * CDIM;
    int tid = threadIdx.x;
    float xv[3];
    float s = 0.f, s2 = 0.f;
#pragma unroll
    for (int i = 0; i < 3; ++i) {
        int c = tid + i * 256;
        float v = a[base + c];
        if (res) v += res[base + c];
        xv[i] = v;
        s += v;
        s2 += v * v;
    }
    rs[tid] = s; rs2[tid] = s2;
    __syncthreads();
    for (int st = 128; st > 0; st >>= 1) {
        if (tid < st) { rs[tid] += rs[tid + st]; rs2[tid] += rs2[tid + st]; }
        __syncthreads();
    }
    float mean = rs[0] * (1.0f / CDIM);
    float var  = rs2[0] * (1.0f / CDIM) - mean * mean;
    float rstd = rsqrtf(var + 1e-5f);
#pragma unroll
    for (int i = 0; i < 3; ++i) {
        int c = tid + i * 256;
        out[base + c] = (xv[i] - mean) * rstd * w[c] + bv[c];
    }
}

// ===========================================================================
// GEMM: C[M,Nout] = act( A[M,K] @ W[K,Nout] + bias )
// f32 in memory, f16 WMMA with f32 accumulation.
// Batch-strided row map: global row g -> b=g/nb, n=g%nb;
//   A row    = b*a_bs + n          (row length K)
//   C row    = b*c_bs + c_off + n  (row length Nout)
// Block: 256 threads (8 waves as 4x2), tile 128x64, K-step 32.
// ===========================================================================
#define GBM 128
#define GBN 64
#define GBK 32
#define GPAD 8

__global__ __launch_bounds__(256) void gemm_kernel(
    const float* __restrict__ A, const float* __restrict__ W,
    const float* __restrict__ bias, float* __restrict__ Cmat,
    int M, int Nout, int K, int nb, int a_bs, int c_bs, int c_off, int act) {
    __shared__ alignas(32) _Float16 As[GBM][GBK + GPAD];
    __shared__ alignas(32) _Float16 Bs[GBN][GBK + GPAD];

    int tid = threadIdx.x;
    int tileM = blockIdx.y * GBM;
    int tileN = blockIdx.x * GBN;
    int wid = tid >> 5, lane = tid & 31;
    int wr = wid & 3, wc = wid >> 2;        // wave grid 4(M) x 2(N)
    int lm = lane & 15, hk = lane >> 4;

    v8f acc00 = {}, acc01 = {}, acc10 = {}, acc11 = {};

    for (int kb = 0; kb < K; kb += GBK) {
        // stage A tile (f32 -> f16), coalesced over k
#pragma unroll
        for (int i = 0; i < (GBM * GBK) / 256; ++i) {
            int idx = i * 256 + tid;
            int r = idx >> 5, kk = idx & 31;
            int g = tileM + r;
            int b = g / nb, n0 = g - b * nb;
            As[r][kk] = (_Float16)A[(size_t)(b * a_bs + n0) * K + kb + kk];
        }
        // stage W tile transposed: Bs[n][k]  (coalesced over n)
#pragma unroll
        for (int i = 0; i < (GBN * GBK) / 256; ++i) {
            int idx = i * 256 + tid;
            int n = idx & 63, kk = idx >> 6;
            Bs[n][kk] = (_Float16)W[(size_t)(kb + kk) * Nout + tileN + n];
        }
        __syncthreads();

        const _Float16* ar0 = &As[wr * 32 + lm][0];
        const _Float16* ar1 = &As[wr * 32 + 16 + lm][0];
        const _Float16* br0 = &Bs[wc * 32 + lm][0];
        const _Float16* br1 = &Bs[wc * 32 + 16 + lm][0];
        union { v16h v; v8h h[2]; } a0, a1, b0, b1;
        a0.h[0] = *(const v8h*)(ar0 + hk * 8);
        a0.h[1] = *(const v8h*)(ar0 + 16 + hk * 8);
        a1.h[0] = *(const v8h*)(ar1 + hk * 8);
        a1.h[1] = *(const v8h*)(ar1 + 16 + hk * 8);
        b0.h[0] = *(const v8h*)(br0 + hk * 8);
        b0.h[1] = *(const v8h*)(br0 + 16 + hk * 8);
        b1.h[0] = *(const v8h*)(br1 + hk * 8);
        b1.h[1] = *(const v8h*)(br1 + 16 + hk * 8);

        acc00 = __builtin_amdgcn_wmma_f32_16x16x32_f16(false, a0.v, false, b0.v, (short)0, acc00, false, false);
        acc01 = __builtin_amdgcn_wmma_f32_16x16x32_f16(false, a0.v, false, b1.v, (short)0, acc01, false, false);
        acc10 = __builtin_amdgcn_wmma_f32_16x16x32_f16(false, a1.v, false, b0.v, (short)0, acc10, false, false);
        acc11 = __builtin_amdgcn_wmma_f32_16x16x32_f16(false, a1.v, false, b1.v, (short)0, acc11, false, false);
        __syncthreads();
    }

    float bv0 = bias ? bias[tileN + wc * 32 + lm] : 0.f;
    float bv1 = bias ? bias[tileN + wc * 32 + 16 + lm] : 0.f;

    auto store_tile = [&](const v8f& acc, int mi, int nj, float bv) {
        int gcol = tileN + wc * 32 + nj * 16 + lm;
#pragma unroll
        for (int v = 0; v < 8; ++v) {
            int gm = tileM + wr * 32 + mi * 16 + v + 8 * hk;
            int b = gm / nb, n0 = gm - b * nb;
            float val = acc[v] + bv;
            if (act == 1) val = gelu_exact(val);
            Cmat[(size_t)(b * c_bs + c_off + n0) * Nout + gcol] = val;
        }
    };
    store_tile(acc00, 0, 0, bv0);
    store_tile(acc01, 0, 1, bv1);
    store_tile(acc10, 1, 0, bv0);
    store_tile(acc11, 1, 1, bv1);
}

// ===========================================================================
// Fused flash-attention: out = softmax(Qh Kh^T / sqrt(128)) Vh   per (b,h)
// grid = (Nq/64, batches*H), block = 128 (4 waves); each wave owns 16 q rows.
// Q/K/V/out are projection buffers [tokens][C]; head h = columns h*128..+127.
// ===========================================================================
__global__ __launch_bounds__(128) void attn_kernel(
    const float* __restrict__ Q, const float* __restrict__ Km,
    const float* __restrict__ Vm, float* __restrict__ O,
    int Nk, int q_bs, int kv_bs) {
    __shared__ alignas(32) _Float16 Qs[64][DHEAD + 8];  // [qrow][d]
    __shared__ alignas(32) _Float16 Kt[32][DHEAD + 8];  // [key][d]   (= Bt for S)
    __shared__ alignas(32) _Float16 Vt[DHEAD][40];      // [d][key]   (= Bt for O)
    __shared__ alignas(32) _Float16 Ps[4][16][40];      // per-wave P bounce

    int tid = threadIdx.x;
    int wid = tid >> 5, lane = tid & 31;
    int lm = lane & 15, hk = lane >> 4;
    int bh = blockIdx.y;
    int b = bh / HEADS, h = bh - b * HEADS;
    int q0r = blockIdx.x * 64;

    const float* qb = Q + (size_t)b * q_bs * CDIM + h * DHEAD;
    const float* kb = Km + (size_t)b * kv_bs * CDIM + h * DHEAD;
    const float* vb = Vm + (size_t)b * kv_bs * CDIM + h * DHEAD;
    float* ob = O + (size_t)b * q_bs * CDIM + h * DHEAD;

    // stage Q tile once
    for (int i = 0; i < 64; ++i) {
        int idx = i * 128 + tid;
        int r = idx >> 7, d = idx & 127;
        Qs[r][d] = (_Float16)qb[(size_t)(q0r + r) * CDIM + d];
    }
    __syncthreads();

    v8f o[8];
#pragma unroll
    for (int t = 0; t < 8; ++t) o[t] = {};
    float mi[8], li[8];
#pragma unroll
    for (int v = 0; v < 8; ++v) { mi[v] = -1e30f; li[v] = 0.f; }
    const float scale = 0.08838834764831845f;  // 1/sqrt(128)

    for (int kb0 = 0; kb0 < Nk; kb0 += 32) {
        __syncthreads();
        for (int i = 0; i < 32; ++i) {
            int idx = i * 128 + tid;
            int kk = idx >> 7, d = idx & 127;
            Kt[kk][d] = (_Float16)kb[(size_t)(kb0 + kk) * CDIM + d];
        }
        for (int i = 0; i < 32; ++i) {
            int idx = i * 128 + tid;
            int kk = idx >> 7, d = idx & 127;
            Vt[d][kk] = (_Float16)vb[(size_t)(kb0 + kk) * CDIM + d];
        }
        __syncthreads();

        if (kb0 + 32 < Nk) {  // prefetch next key block (global_prefetch_b8)
            __builtin_prefetch(&kb[(size_t)(kb0 + 32 + (tid >> 2)) * CDIM + (tid & 3) * 32], 0, 1);
            __builtin_prefetch(&vb[(size_t)(kb0 + 32 + (tid >> 2)) * CDIM + (tid & 3) * 32], 0, 1);
        }

        // ---- S = Q K^T : two 16x16 key tiles, K'=128 in 4 steps of 32 ----
        v8f s0 = {}, s1 = {};
        const _Float16* qrow = &Qs[wid * 16 + lm][0];
#pragma unroll
        for (int ks = 0; ks < 4; ++ks) {
            union { v16h v; v8h h[2]; } af, bf0, bf1;
            af.h[0] = *(const v8h*)(qrow + ks * 32 + hk * 8);
            af.h[1] = *(const v8h*)(qrow + ks * 32 + 16 + hk * 8);
            bf0.h[0] = *(const v8h*)(&Kt[lm][ks * 32 + hk * 8]);
            bf0.h[1] = *(const v8h*)(&Kt[lm][ks * 32 + 16 + hk * 8]);
            bf1.h[0] = *(const v8h*)(&Kt[16 + lm][ks * 32 + hk * 8]);
            bf1.h[1] = *(const v8h*)(&Kt[16 + lm][ks * 32 + 16 + hk * 8]);
            s0 = __builtin_amdgcn_wmma_f32_16x16x32_f16(false, af.v, false, bf0.v, (short)0, s0, false, false);
            s1 = __builtin_amdgcn_wmma_f32_16x16x32_f16(false, af.v, false, bf1.v, (short)0, s1, false, false);
        }

        // ---- online softmax update; lane owns rows m = v + 8*hk ----
#pragma unroll
        for (int v = 0; v < 8; ++v) {
            float a0 = s0[v] * scale;
            float a1 = s1[v] * scale;
            float x = fmaxf(a0, a1);
            for (int msk = 8; msk >= 1; msk >>= 1) x = fmaxf(x, __shfl_xor(x, msk, 32));
            float mnew = fmaxf(mi[v], x);
            float alpha = __expf(mi[v] - mnew);
            float p0 = __expf(a0 - mnew);
            float p1 = __expf(a1 - mnew);
            float rs = p0 + p1;
            for (int msk = 8; msk >= 1; msk >>= 1) rs += __shfl_xor(rs, msk, 32);
            li[v] = li[v] * alpha + rs;
            mi[v] = mnew;
#pragma unroll
            for (int t = 0; t < 8; ++t) o[t][v] = o[t][v] * alpha;
            Ps[wid][8 * hk + v][lm] = (_Float16)p0;
            Ps[wid][8 * hk + v][16 + lm] = (_Float16)p1;
        }

        // ---- O += P @ V : A-frag from wave-private LDS bounce ----
        const _Float16* prow = &Ps[wid][lm][0];
        union { v16h v; v8h h[2]; } pf;
        pf.h[0] = *(const v8h*)(prow + hk * 8);
        pf.h[1] = *(const v8h*)(prow + 16 + hk * 8);
#pragma unroll
        for (int t = 0; t < 8; ++t) {
            union { v16h v; v8h h[2]; } vf;
            vf.h[0] = *(const v8h*)(&Vt[t * 16 + lm][hk * 8]);
            vf.h[1] = *(const v8h*)(&Vt[t * 16 + lm][16 + hk * 8]);
            o[t] = __builtin_amdgcn_wmma_f32_16x16x32_f16(false, pf.v, false, vf.v, (short)0, o[t], false, false);
        }
    }

    // ---- finalize: O /= l ----
#pragma unroll
    for (int v = 0; v < 8; ++v) {
        float inv = 1.0f / li[v];
        int gr = q0r + wid * 16 + 8 * hk + v;
#pragma unroll
        for (int t = 0; t < 8; ++t) {
            ob[(size_t)gr * CDIM + t * 16 + lm] = o[t][v] * inv;
        }
    }
}

// ===========================================================================
// Host-side orchestration
// ===========================================================================
static inline void launch_gemm(const float* A, const float* W, const float* bias,
                               float* C, int M, int Nout, int K,
                               int nb, int a_bs, int c_bs, int c_off, int act,
                               hipStream_t s) {
    dim3 grid(Nout / GBN, M / GBM);
    gemm_kernel<<<grid, 256, 0, s>>>(A, W, bias, C, M, Nout, K, nb, a_bs, c_bs, c_off, act);
}

static inline void launch_prep(float* out, const float* x, const float* y,
                               const float* f, int rows, int nb, int ldo, int off,
                               hipStream_t s) {
    prep_kernel<<<rows * 3, 256, 0, s>>>(out, x, y, f, nb, ldo, off);
}

extern "C" void kernel_launch(void* const* d_in, const int* in_sizes, int n_in,
                              void* d_out, int out_size, void* d_ws, size_t ws_size,
                              hipStream_t stream) {
    (void)in_sizes; (void)n_in; (void)out_size; (void)ws_size;
    const float* queries    = (const float*)d_in[0];
    const float* change_emb = (const float*)d_in[1];
    const float* imageA_emb = (const float*)d_in[2];
    const float* imageB_emb = (const float*)d_in[3];
    const float* pe         = (const float*)d_in[4];
    const float* wq1 = (const float*)d_in[5];  const float* bq1 = (const float*)d_in[6];
    const float* wk1 = (const float*)d_in[7];  const float* bk1 = (const float*)d_in[8];
    const float* wv1 = (const float*)d_in[9];  const float* bv1 = (const float*)d_in[10];
    const float* wo1 = (const float*)d_in[11]; const float* bo1 = (const float*)d_in[12];
    const float* wq2 = (const float*)d_in[13]; const float* bq2 = (const float*)d_in[14];
    const float* wk2 = (const float*)d_in[15]; const float* bk2 = (const float*)d_in[16];
    const float* wv2 = (const float*)d_in[17]; const float* bv2 = (const float*)d_in[18];
    const float* wo2 = (const float*)d_in[19]; const float* bo2 = (const float*)d_in[20];
    const float* wq3 = (const float*)d_in[21]; const float* bq3 = (const float*)d_in[22];
    const float* wk3 = (const float*)d_in[23]; const float* bk3 = (const float*)d_in[24];
    const float* wv3 = (const float*)d_in[25]; const float* bv3 = (const float*)d_in[26];
    const float* wo3 = (const float*)d_in[27]; const float* bo3 = (const float*)d_in[28];
    const float* ln1w = (const float*)d_in[29]; const float* ln1b = (const float*)d_in[30];
    const float* ln2w = (const float*)d_in[31]; const float* ln2b = (const float*)d_in[32];
    const float* ln3w = (const float*)d_in[33]; const float* ln3b = (const float*)d_in[34];
    const float* mw1 = (const float*)d_in[35]; const float* mb1 = (const float*)d_in[36];
    const float* mw2 = (const float*)d_in[37]; const float* mb2 = (const float*)d_in[38];
    const float* flag_a = (const float*)d_in[39];
    const float* flag_b = (const float*)d_in[40];
    const float* flag_c = (const float*)d_in[41];

    float* out = (float*)d_out;
    float* ws = (float*)d_ws;
    const size_t S1 = (size_t)4096 * CDIM;  // B*N*C tokens block

    // bump allocation with lifetime aliasing (total 18*S1 floats ~227MB)
    float* q02 = ws;            // q0, later q2
    float* q13 = ws + S1;       // q1, later q3 (= change_out)
    float* X   = ws + 2 * S1;   // staging, up to 8192 token rows
    float* PQ  = ws + 4 * S1;   // Q projections, up to 8192 rows
    float* PK  = ws + 6 * S1;   // K projections, up to 16384 rows (also MLP hidden)
    float* PV  = ws + 10 * S1;  // V projections, up to 16384 rows
    float* CTX = ws + 14 * S1;  // attention context / staging, up to 8192 rows
    float* AO  = ws + 16 * S1;  // output-projection result, up to 8192 rows

    // ---------------- Self-attention on change queries --------------------
    launch_prep(q02, queries, change_emb, nullptr, 4096, 4096, 4096, 0, stream);  // q0
    launch_prep(X, q02, pe, nullptr, 4096, 4096, 4096, 0, stream);                // q0+pe
    launch_gemm(X, wq1, bq1, PQ, 4096, CDIM, CDIM, 4096, 4096, 4096, 0, 0, stream);
    launch_gemm(X, wk1, bk1, PK, 4096, CDIM, CDIM, 4096, 4096, 4096, 0, 0, stream);
    launch_gemm(q02, wv1, bv1, PV, 4096, CDIM, CDIM, 4096, 4096, 4096, 0, 0, stream);
    attn_kernel<<<dim3(16, 4 * HEADS), 128, 0, stream>>>(PQ, PK, PV, CTX, 1024, 1024, 1024);
    launch_gemm(CTX, wo1, bo1, AO, 4096, CDIM, CDIM, 4096, 4096, 4096, 0, 0, stream);
    ln_kernel<<<4096, 256, 0, stream>>>(q13, q02, AO, ln1w, ln1b);  // q1

    // ---------------- Cross-attention: change -> [A; B; change] -----------
    launch_prep(X, q13, pe, flag_c, 4096, 4096, 4096, 0, stream);  // qf
    launch_gemm(X, wq2, bq2, PQ, 4096, CDIM, CDIM, 4096, 4096, 4096, 0, 0, stream);
    launch_prep(X, imageA_emb, pe, flag_a, 4096, 4096, 4096, 0, stream);
    launch_gemm(X, wk2, bk2, PK, 4096, CDIM, CDIM, 1024, 1024, 3072, 0, 0, stream);
    launch_prep(X, imageB_emb, pe, flag_b, 4096, 4096, 4096, 0, stream);
    launch_gemm(X, wk2, bk2, PK, 4096, CDIM, CDIM, 1024, 1024, 3072, 1024, 0, stream);
    launch_prep(X, q13, pe, flag_c, 4096, 4096, 4096, 0, stream);
    launch_gemm(X, wk2, bk2, PK, 4096, CDIM, CDIM, 1024, 1024, 3072, 2048, 0, stream);
    launch_gemm(imageA_emb, wv2, bv2, PV, 4096, CDIM, CDIM, 1024, 1024, 3072, 0, 0, stream);
    launch_gemm(imageB_emb, wv2, bv2, PV, 4096, CDIM, CDIM, 1024, 1024, 3072, 1024, 0, stream);
    launch_gemm(q13, wv2, bv2, PV, 4096, CDIM, CDIM, 1024, 1024, 3072, 2048, 0, stream);
    attn_kernel<<<dim3(16, 4 * HEADS), 128, 0, stream>>>(PQ, PK, PV, CTX, 3072, 1024, 3072);
    launch_gemm(CTX, wo2, bo2, AO, 4096, CDIM, CDIM, 4096, 4096, 4096, 0, 0, stream);
    ln_kernel<<<4096, 256, 0, stream>>>(q02, q13, AO, ln2w, ln2b);  // q2

    // ---------------- MLP (exact GELU) -------------------------------------
    launch_gemm(q02, mw1, mb1, PK /*hidden*/, 4096, 2304, CDIM, 4096, 4096, 4096, 0, 1, stream);
    launch_gemm(PK, mw2, mb2, CTX, 4096, CDIM, 2304, 4096, 4096, 4096, 0, 0, stream);
    ln_kernel<<<4096, 256, 0, stream>>>(q13, q02, CTX, ln3w, ln3b);  // q3 = change_out
    launch_prep(out + 2 * S1, q13, nullptr, nullptr, 4096, 4096, 4096, 0, stream);

    // ---------------- Cross-attention: [A;B] -> [image; change] -----------
    launch_prep(X, imageA_emb, pe, flag_a, 4096, 4096, 4096, 0, stream);             // qa
    launch_prep(X + 4 * 1024 * CDIM, imageB_emb, pe, flag_b, 4096, 4096, 4096, 0, stream);  // qb
    launch_gemm(X, wq3, bq3, PQ, 8192, CDIM, CDIM, 8192, 8192, 8192, 0, 0, stream);
    // K: batches 0-3 seg0 = qa rows; batches 4-7 seg0 = qb rows
    launch_gemm(X, wk3, bk3, PK, 4096, CDIM, CDIM, 1024, 1024, 2048, 0, 0, stream);
    launch_gemm(X + 4 * 1024 * CDIM, wk3, bk3, PK + (size_t)4 * 2048 * CDIM,
                4096, CDIM, CDIM, 1024, 1024, 2048, 0, 0, stream);
    launch_prep(CTX, q13, pe, flag_c, 4096, 4096, 4096, 0, stream);  // q3+pe+flag_c
    launch_gemm(CTX, wk3, bk3, PK, 4096, CDIM, CDIM, 1024, 1024, 2048, 1024, 0, stream);
    launch_gemm(CTX, wk3, bk3, PK + (size_t)4 * 2048 * CDIM,
                4096, CDIM, CDIM, 1024, 1024, 2048, 1024, 0, stream);
    // V: raw buffers, no adds
    launch_gemm(imageA_emb, wv3, bv3, PV, 4096, CDIM, CDIM, 1024, 1024, 2048, 0, 0, stream);
    launch_gemm(imageB_emb, wv3, bv3, PV + (size_t)4 * 2048 * CDIM,
                4096, CDIM, CDIM, 1024, 1024, 2048, 0, 0, stream);
    launch_gemm(q13, wv3, bv3, PV, 4096, CDIM, CDIM, 1024, 1024, 2048, 1024, 0, stream);
    launch_gemm(q13, wv3, bv3, PV + (size_t)4 * 2048 * CDIM,
                4096, CDIM, CDIM, 1024, 1024, 2048, 1024, 0, stream);
    attn_kernel<<<dim3(16, 8 * HEADS), 128, 0, stream>>>(PQ, PK, PV, CTX, 2048, 1024, 2048);
    launch_gemm(CTX, wo3, bo3, AO, 8192, CDIM, CDIM, 8192, 8192, 8192, 0, 0, stream);
    // final residual: out[0:2*S1] = (qa;qb) + attn  -> imageA_refined, imageB_refined
    launch_prep(out, X, AO, nullptr, 8192, 8192, 8192, 0, stream);
}
